// ViT_66571993088920
// MI455X (gfx1250) — matrix operand order repaired and checked by reading, not compiled, not run on hardware
//
#include <hip/hip_runtime.h>
#include <hip/hip_bf16.h>

// ---------------------------------------------------------------------------
// ViT forward for MI455X (gfx1250, wave32, WMMA).
// B=32, C=3, IMG=384, P=16, D=512, HEADS=4, DH=128, L=6, DFF=2048, NCLS=10
// All GEMMs: bf16 operands, fp32 accumulate via v_wmma_f32_16x16x32_bf16.
// GEMM A-tiles staged in LDS via async global->LDS (ASYNCcnt) when available.
// ---------------------------------------------------------------------------

typedef __bf16 bf16_t;
typedef __attribute__((ext_vector_type(16))) __bf16 v16bf;
typedef __attribute__((ext_vector_type(8)))  float  v8f;
typedef __attribute__((ext_vector_type(8)))  int    v8i;
typedef __attribute__((ext_vector_type(4)))  int    v4i;

#define BATCH 32
#define TOK   577
#define TPATCH 576
#define DMODEL 512
#define NHEAD 4
#define DH    128
#define NLAYER 6
#define DFF   2048
#define NCLS  10
#define MTOK  (BATCH * TOK)      // 18464
#define MPAT  (BATCH * TPATCH)   // 18432
#define KPAT  768                // P*P*C
#define TPAD  608                // 19 * 32 key padding for attention
#define QTILES 37                // ceil(577/16)

// ---- async global->LDS support (gfx1250 GLOBAL_LOAD_ASYNC_TO_LDS_B128) ----
#if __has_builtin(__builtin_amdgcn_global_load_async_to_lds_b128)
#define HAVE_ASYNC_LDS 1
#endif

#if defined(HAVE_ASYNC_LDS) && __has_builtin(__builtin_amdgcn_s_wait_asynccnt)
#define WAIT_ASYNC() __builtin_amdgcn_s_wait_asynccnt(0)
#elif defined(HAVE_ASYNC_LDS)
#define WAIT_ASYNC() asm volatile("s_wait_asynccnt 0" ::: "memory")
#else
#define WAIT_ASYNC()
#endif

// Builtin is strongly typed: b128 variant takes v4i* in AS1 (global) / AS3 (LDS).
typedef __attribute__((address_space(1))) v4i g_v4i;
typedef __attribute__((address_space(3))) v4i l_v4i;

__device__ __forceinline__ v8f zero8() {
  v8f z = {0.f, 0.f, 0.f, 0.f, 0.f, 0.f, 0.f, 0.f};
  return z;
}

__device__ __forceinline__ v8f wmma_bf16(v16bf a, v16bf b, v8f c) {
  return __builtin_amdgcn_wmma_f32_16x16x32_bf16(
      false, a, false, b, (short)0, c, false, false);
}

// A-fragment (16x32 bf16, MxK) from row-major global memory (ISA 7.12.2).
__device__ __forceinline__ v16bf load_a_frag(const bf16_t* __restrict__ A,
                                             int lda, int m0, int k0,
                                             int ln, int half) {
  const int* p = reinterpret_cast<const int*>(
      A + (size_t)(m0 + ln) * lda + k0 + half * 16);
  v8i t;
  t[0] = p[0];  t[1] = p[1];  t[2] = p[2];  t[3] = p[3];
  t[4] = p[4];  t[5] = p[5];  t[6] = p[6];  t[7] = p[7];
  return __builtin_bit_cast(v16bf, t);
}

// B-fragment (32x16 bf16, KxN) from a transposed (NxK) matrix.
__device__ __forceinline__ v16bf load_b_frag(const bf16_t* __restrict__ Bt,
                                             int ldb, int n0, int k0,
                                             int ln, int half) {
  return *reinterpret_cast<const v16bf*>(
      Bt + (size_t)(n0 + ln) * ldb + k0 + half * 16);
}

// A-fragment read from an LDS tile with 80-byte (40-element) row stride.
__device__ __forceinline__ v16bf lds_a_frag(const bf16_t* buf, int row,
                                            int half) {
  const v4i* p = reinterpret_cast<const v4i*>(
      reinterpret_cast<const char*>(buf) + row * 80 + half * 32);
  v4i lo = p[0];  // dwords 0..3
  v4i hi = p[1];  // dwords 4..7
  v8i t;
#pragma unroll
  for (int i = 0; i < 4; ++i) { t[i] = lo[i]; t[i + 4] = hi[i]; }
  return __builtin_bit_cast(v16bf, t);
}

// Stage a 64-row x 32-k A tile into LDS (one b128 per thread).
__device__ __forceinline__ void stage_a_tile(const bf16_t* __restrict__ A,
                                             int M, int K, int mblk, int k0,
                                             bf16_t* lbuf) {
  int t = threadIdx.x;
  int row = t >> 2, chunk = t & 3;
  int grow = mblk * 64 + row;
  if (grow > M - 1) grow = M - 1;  // clamp (partial last tile)
  const bf16_t* gsrc = A + (size_t)grow * K + k0 + chunk * 8;
  bf16_t* ldst = lbuf + row * 40 + chunk * 8;
#ifdef HAVE_ASYNC_LDS
  __builtin_amdgcn_global_load_async_to_lds_b128(
      (g_v4i*)(uintptr_t)gsrc, (l_v4i*)(uintptr_t)ldst, 0, 0);
#else
  *reinterpret_cast<v4i*>(ldst) = *reinterpret_cast<const v4i*>(gsrc);
#endif
}

__device__ __forceinline__ float rmax16(float v) {
#pragma unroll
  for (int m = 8; m >= 1; m >>= 1) v = fmaxf(v, __shfl_xor(v, m, 16));
  return v;
}
__device__ __forceinline__ float rsum16(float v) {
#pragma unroll
  for (int m = 8; m >= 1; m >>= 1) v += __shfl_xor(v, m, 16);
  return v;
}

// ---------------------------------------------------------------------------
// Elementwise / prep kernels
// ---------------------------------------------------------------------------

__global__ void wconv_kernel(const float* __restrict__ W,
                             bf16_t* __restrict__ Wt, int K, int N) {
  int idx = blockIdx.x * 256 + threadIdx.x;
  if (idx >= K * N) return;
  int n = idx / K;
  int k = idx - n * K;
  Wt[idx] = (bf16_t)W[(size_t)k * N + n];
}

__global__ void patch_gather_kernel(const float* __restrict__ img,
                                    bf16_t* __restrict__ Apat) {
  size_t idx = (size_t)blockIdx.x * 256 + threadIdx.x;  // MPAT*KPAT
  int k = (int)(idx % KPAT);
  size_t bt = idx / KPAT;
  int t = (int)(bt % TPATCH);
  int b = (int)(bt / TPATCH);
  int c = k % 3;
  int py = (k / 3) % 16;
  int px = k / 48;
  int x = t / 24;
  int y = t - x * 24;
  size_t src = ((size_t)(b * 3 + c) * 384 + (px * 24 + x)) * 384 + (py * 24 + y);
  Apat[idx] = (bf16_t)img[src];
}

__global__ void assemble_tok_kernel(const float* __restrict__ ptmp,
                                    const float* __restrict__ cls,
                                    const float* __restrict__ pos,
                                    float* __restrict__ x,
                                    bf16_t* __restrict__ xbf) {
  size_t idx = (size_t)blockIdx.x * 256 + threadIdx.x;  // MTOK*DMODEL
  int d = (int)(idx & (DMODEL - 1));
  size_t rt = idx >> 9;
  int t = (int)(rt % TOK);
  int b = (int)(rt / TOK);
  float v = (t == 0) ? cls[d]
                     : ptmp[((size_t)b * TPATCH + (t - 1)) * DMODEL + d];
  v += pos[(size_t)t * DMODEL + d];
  x[idx] = v;
  xbf[idx] = (bf16_t)v;
}

__global__ void split_qkv_kernel(const bf16_t* __restrict__ qkv,
                                 bf16_t* __restrict__ Qp,
                                 bf16_t* __restrict__ Kp,
                                 bf16_t* __restrict__ Vp) {
  size_t idx = (size_t)blockIdx.x * 256 + threadIdx.x;  // 128*TPAD*DH
  int dh = (int)(idx & (DH - 1));
  size_t r = idx >> 7;
  int t = (int)(r % TPAD);
  int bh = (int)(r / TPAD);
  int b = bh >> 2, h = bh & 3;
  bf16_t q = (bf16_t)0.0f, k = (bf16_t)0.0f, v = (bf16_t)0.0f;
  if (t < TOK) {
    const bf16_t* p = qkv + ((size_t)(b * TOK + t)) * (3 * DMODEL) + dh * 12 + h;
    q = p[0];
    k = p[4];
    v = p[8];
  }
  Qp[idx] = q;
  Kp[idx] = k;
  Vp[((size_t)bh * DH + dh) * TPAD + t] = v;
}

__global__ void add_ln_kernel(const float* __restrict__ a,
                              const float* __restrict__ res,
                              const float* __restrict__ g,
                              const float* __restrict__ bb,
                              float* __restrict__ yf,
                              bf16_t* __restrict__ ybf) {
  int row = blockIdx.x;
  int tid = threadIdx.x;  // 256 threads, 2 elems each
  const float* ar = a + (size_t)row * DMODEL;
  const float* rr = res + (size_t)row * DMODEL;
  float v0 = ar[tid] + rr[tid];
  float v1 = ar[tid + 256] + rr[tid + 256];
  float s1 = v0 + v1;
  float s2 = v0 * v0 + v1 * v1;
#pragma unroll
  for (int m = 16; m >= 1; m >>= 1) {
    s1 += __shfl_xor(s1, m, 32);
    s2 += __shfl_xor(s2, m, 32);
  }
  __shared__ float red[2][8];
  int w = tid >> 5;
  if ((tid & 31) == 0) { red[0][w] = s1; red[1][w] = s2; }
  __syncthreads();
  float S1 = 0.f, S2 = 0.f;
#pragma unroll
  for (int i = 0; i < 8; ++i) { S1 += red[0][i]; S2 += red[1][i]; }
  float mean = S1 * (1.0f / DMODEL);
  float var = S2 * (1.0f / DMODEL) - mean * mean;
  float rstd = rsqrtf(var + 1e-5f);
  float o0 = (v0 - mean) * rstd * g[tid] + bb[tid];
  float o1 = (v1 - mean) * rstd * g[tid + 256] + bb[tid + 256];
  size_t base = (size_t)row * DMODEL;
  yf[base + tid] = o0;
  yf[base + tid + 256] = o1;
  ybf[base + tid] = (bf16_t)o0;
  ybf[base + tid + 256] = (bf16_t)o1;
}

__global__ void head_kernel(const float* __restrict__ x,
                            const float* __restrict__ Wh,
                            const float* __restrict__ bh,
                            float* __restrict__ out) {
  int tid = threadIdx.x;
  if (tid >= BATCH * NCLS) return;
  int b = tid / NCLS, c = tid - b * NCLS;
  const float* xr = x + (size_t)b * TOK * DMODEL;
  float s = bh[c];
  for (int k = 0; k < DMODEL; ++k) s += xr[k] * Wh[k * NCLS + c];
  out[tid] = s;
}

// ---------------------------------------------------------------------------
// WMMA GEMM: C[M,N] = A[M,K](bf16) * Wt[N,K](bf16)^T (+bias)(+gelu)
// Block = 256 threads = 8 waves as 2(M) x 4(N); block tile 64 x 128.
// Each wave: 32x32 output (2x2 WMMA tiles). A tile double-buffered in LDS
// via async global->LDS; B (weights) streamed from global with prefetch.
// Requires: N % 128 == 0, K % 32 == 0. Partial M tile handled by clamp+guard.
// ---------------------------------------------------------------------------
template <int OP>  // 0 = none, 1 = exact GELU
__global__ __launch_bounds__(256, 1) void gemm_bf16_kernel(
    const bf16_t* __restrict__ A, const bf16_t* __restrict__ Wt,
    const float* __restrict__ bias, float* __restrict__ Cf,
    bf16_t* __restrict__ Cbf, int M, int N, int K) {
  __shared__ __align__(16) bf16_t lds_a[2][64 * 40];  // 80B row stride
  int mblk = blockIdx.y, nblk = blockIdx.x;
  int w = threadIdx.x >> 5;
  int mw = w >> 2, nw = w & 3;
  int lane = threadIdx.x & 31, ln = lane & 15, half = lane >> 4;
  int row0 = mw * 32;                    // within block tile
  int col0 = nblk * 128 + nw * 32;       // global column

  v8f acc[2][2];
  acc[0][0] = zero8(); acc[0][1] = zero8();
  acc[1][0] = zero8(); acc[1][1] = zero8();

  stage_a_tile(A, M, K, mblk, 0, lds_a[0]);
  int nsteps = K >> 5;
  for (int s = 0; s < nsteps; ++s) {
    int k0 = s << 5;
    WAIT_ASYNC();
    __syncthreads();
    int cur = s & 1;
    if (s + 1 < nsteps) stage_a_tile(A, M, K, mblk, k0 + 32, lds_a[cur ^ 1]);
    if (s + 2 < nsteps) {
      __builtin_prefetch(Wt + (size_t)(col0 + ln) * K + k0 + 64, 0, 1);
      __builtin_prefetch(Wt + (size_t)(col0 + 16 + ln) * K + k0 + 64, 0, 1);
    }
    v16bf a0 = lds_a_frag(lds_a[cur], row0 + ln, half);
    v16bf a1 = lds_a_frag(lds_a[cur], row0 + 16 + ln, half);
    v16bf b0 = load_b_frag(Wt, K, col0, k0, ln, half);
    v16bf b1 = load_b_frag(Wt, K, col0 + 16, k0, ln, half);
    acc[0][0] = wmma_bf16(a0, b0, acc[0][0]);
    acc[0][1] = wmma_bf16(a0, b1, acc[0][1]);
    acc[1][0] = wmma_bf16(a1, b0, acc[1][0]);
    acc[1][1] = wmma_bf16(a1, b1, acc[1][1]);
  }
#pragma unroll
  for (int i = 0; i < 2; ++i) {
#pragma unroll
    for (int j = 0; j < 2; ++j) {
      int col = col0 + j * 16 + ln;
      float bv = bias ? bias[col] : 0.f;
#pragma unroll
      for (int r = 0; r < 8; ++r) {
        int grow = mblk * 64 + row0 + i * 16 + r + 8 * half;
        if (grow < M) {
          float v = acc[i][j][r] + bv;
          if (OP == 1) v = 0.5f * v * (1.0f + erff(v * 0.70710678118654752f));
          size_t o = (size_t)grow * N + col;
          if (Cf) Cf[o] = v;
          if (Cbf) Cbf[o] = (bf16_t)v;
        }
      }
    }
  }
}

// ---------------------------------------------------------------------------
// Flash attention: one wave = one (b,h, 16-query tile); 32 keys per step.
// ---------------------------------------------------------------------------
__global__ __launch_bounds__(256, 1) void flash_attn_kernel(
    const bf16_t* __restrict__ Q, const bf16_t* __restrict__ Kt,
    const bf16_t* __restrict__ V, bf16_t* __restrict__ Obf) {
  int wid = blockIdx.x * 8 + (threadIdx.x >> 5);  // exactly 128*37 waves
  int bh = wid / QTILES;
  int qt = wid - bh * QTILES;
  int b = bh >> 2, h = bh & 3;
  int lane = threadIdx.x & 31;
  int ln = lane & 15, half = lane >> 4;

  const bf16_t* Qb = Q + (size_t)bh * TPAD * DH;
  const bf16_t* Kb = Kt + (size_t)bh * TPAD * DH;
  const bf16_t* Vb = V + (size_t)bh * DH * TPAD;

  v16bf qf[4];
#pragma unroll
  for (int kk = 0; kk < 4; ++kk)
    qf[kk] = load_a_frag(Qb, DH, qt * 16, kk * 32, ln, half);

  v8f oacc[8];
#pragma unroll
  for (int c = 0; c < 8; ++c) oacc[c] = zero8();
  float mrow[8], lrow[8];
#pragma unroll
  for (int r = 0; r < 8; ++r) { mrow[r] = -1e30f; lrow[r] = 0.f; }

  __shared__ __align__(16) bf16_t lds_p[8][16 * 32];
  bf16_t* myp = &lds_p[threadIdx.x >> 5][0];

  const float sc = 0.08838834764831845f;  // DH^-0.5
  for (int j0 = 0; j0 < TPAD; j0 += 32) {
    v8f sacc[2];
    sacc[0] = zero8();
    sacc[1] = zero8();
#pragma unroll
    for (int kk = 0; kk < 4; ++kk) {
      v16bf b0 = load_b_frag(Kb, DH, j0, kk * 32, ln, half);
      v16bf b1 = load_b_frag(Kb, DH, j0 + 16, kk * 32, ln, half);
      sacc[0] = wmma_bf16(qf[kk], b0, sacc[0]);
      sacc[1] = wmma_bf16(qf[kk], b1, sacc[1]);
    }
    bool valid0 = (j0 + ln) < TOK;
    bool valid1 = (j0 + 16 + ln) < TOK;
#pragma unroll
    for (int r = 0; r < 8; ++r) {
      float s0 = valid0 ? sacc[0][r] * sc : -1e30f;
      float s1 = valid1 ? sacc[1][r] * sc : -1e30f;
      float rm = rmax16(fmaxf(s0, s1));
      float mn = fmaxf(mrow[r], rm);
      float alpha = __expf(mrow[r] - mn);
      float p0 = __expf(s0 - mn);  // masked lanes -> 0
      float p1 = __expf(s1 - mn);
      float rs = rsum16(p0 + p1);
      lrow[r] = lrow[r] * alpha + rs;
      mrow[r] = mn;
#pragma unroll
      for (int c = 0; c < 8; ++c) oacc[c][r] *= alpha;
      myp[(r + 8 * half) * 32 + ln] = (bf16_t)p0;
      myp[(r + 8 * half) * 32 + 16 + ln] = (bf16_t)p1;
    }
    __syncthreads();
    const int* pp = reinterpret_cast<const int*>(myp + ln * 32 + half * 16);
    v8i pi;
    pi[0] = pp[0]; pi[1] = pp[1]; pi[2] = pp[2];  pi[3] = pp[3];
    pi[4] = pp[4]; pi[5] = pp[5]; pi[6] = pp[6]; pi[7] = pp[7];
    v16bf pf = __builtin_bit_cast(v16bf, pi);
    __syncthreads();
#pragma unroll
    for (int c = 0; c < 8; ++c) {
      v16bf bf = *reinterpret_cast<const v16bf*>(
          Vb + (size_t)(c * 16 + ln) * TPAD + j0 + half * 16);
      oacc[c] = wmma_bf16(pf, bf, oacc[c]);
    }
  }

#pragma unroll
  for (int r = 0; r < 8; ++r) {
    int qrow = qt * 16 + r + 8 * half;
    if (qrow < TOK) {
      float inv = 1.0f / lrow[r];
      size_t base = ((size_t)b * TOK + qrow) * DMODEL + h * DH;
#pragma unroll
      for (int c = 0; c < 8; ++c)
        Obf[base + c * 16 + ln] = (bf16_t)(oacc[c][r] * inv);
    }
  }
}

// ---------------------------------------------------------------------------
// Host orchestration
// ---------------------------------------------------------------------------
static inline size_t algn(size_t x) { return (x + 255) & ~(size_t)255; }

template <int OP>
static void launch_gemm(const bf16_t* A, const bf16_t* Wt, const float* bias,
                        float* Cf, bf16_t* Cbf, int M, int N, int K,
                        hipStream_t s) {
  dim3 grid(N / 128, (M + 63) / 64);
  gemm_bf16_kernel<OP><<<grid, 256, 0, s>>>(A, Wt, bias, Cf, Cbf, M, N, K);
}

static void launch_wconv(const float* W, bf16_t* Wt, int K, int N,
                         hipStream_t s) {
  int n = K * N;
  wconv_kernel<<<(n + 255) / 256, 256, 0, s>>>(W, Wt, K, N);
}

extern "C" void kernel_launch(void* const* d_in, const int* in_sizes, int n_in,
                              void* d_out, int out_size, void* d_ws,
                              size_t ws_size, hipStream_t stream) {
  (void)in_sizes; (void)n_in; (void)out_size; (void)ws_size;
  const float* img    = (const float*)d_in[0];
  const float* Wpatch = (const float*)d_in[1];
  const float* bpatch = (const float*)d_in[2];
  const float* cls    = (const float*)d_in[3];
  const float* pos    = (const float*)d_in[4];
  const float* Wqkv   = (const float*)d_in[5];
  const float* W0     = (const float*)d_in[6];
  const float* g1     = (const float*)d_in[7];
  const float* b1     = (const float*)d_in[8];
  const float* g2     = (const float*)d_in[9];
  const float* b2     = (const float*)d_in[10];
  const float* W1     = (const float*)d_in[11];
  const float* bb1    = (const float*)d_in[12];
  const float* W2     = (const float*)d_in[13];
  const float* bb2    = (const float*)d_in[14];
  const float* Whead  = (const float*)d_in[15];
  const float* bhead  = (const float*)d_in[16];
  float* out = (float*)d_out;

  char* base = (char*)d_ws;
  size_t off = 0;
  auto alloc = [&](size_t bytes) -> void* {
    void* r = base + off;
    off = algn(off + bytes);
    return r;
  };

  const size_t BH = BATCH * NHEAD;  // 128
  bf16_t* wpatcht = (bf16_t*)alloc((size_t)DMODEL * KPAT * 2);
  bf16_t* wqkvt   = (bf16_t*)alloc((size_t)NLAYER * 3 * DMODEL * DMODEL * 2);
  bf16_t* w0t     = (bf16_t*)alloc((size_t)NLAYER * DMODEL * DMODEL * 2);
  bf16_t* w1t     = (bf16_t*)alloc((size_t)NLAYER * DFF * DMODEL * 2);
  bf16_t* w2t     = (bf16_t*)alloc((size_t)NLAYER * DMODEL * DFF * 2);
  bf16_t* apat    = (bf16_t*)alloc((size_t)MPAT * KPAT * 2);
  float*  ptmp    = (float*)alloc((size_t)MPAT * DMODEL * 4);
  float*  x       = (float*)alloc((size_t)MTOK * DMODEL * 4);
  bf16_t* xbf     = (bf16_t*)alloc((size_t)MTOK * DMODEL * 2);
  float*  y       = (float*)alloc((size_t)MTOK * DMODEL * 4);
  bf16_t* ybf     = (bf16_t*)alloc((size_t)MTOK * DMODEL * 2);
  float*  gf32    = (float*)alloc((size_t)MTOK * DMODEL * 4);
  bf16_t* qkvbf   = (bf16_t*)alloc((size_t)MTOK * 3 * DMODEL * 2);
  bf16_t* Qp      = (bf16_t*)alloc(BH * TPAD * DH * 2);
  bf16_t* Kp      = (bf16_t*)alloc(BH * TPAD * DH * 2);
  bf16_t* Vp      = (bf16_t*)alloc(BH * DH * TPAD * 2);
  bf16_t* obf     = (bf16_t*)alloc((size_t)MTOK * DMODEL * 2);
  bf16_t* hbf     = (bf16_t*)alloc((size_t)MTOK * DFF * 2);

  // --- weight conversion (fp32 -> transposed bf16) ---
  launch_wconv(Wpatch, wpatcht, KPAT, DMODEL, stream);
  for (int l = 0; l < NLAYER; ++l) {
    launch_wconv(Wqkv + (size_t)l * DMODEL * 3 * DMODEL,
                 wqkvt + (size_t)l * 3 * DMODEL * DMODEL, DMODEL, 3 * DMODEL,
                 stream);
    launch_wconv(W0 + (size_t)l * DMODEL * DMODEL,
                 w0t + (size_t)l * DMODEL * DMODEL, DMODEL, DMODEL, stream);
    launch_wconv(W1 + (size_t)l * DMODEL * DFF,
                 w1t + (size_t)l * DFF * DMODEL, DMODEL, DFF, stream);
    launch_wconv(W2 + (size_t)l * DFF * DMODEL,
                 w2t + (size_t)l * DMODEL * DFF, DFF, DMODEL, stream);
  }

  // --- patch embedding ---
  {
    size_t n = (size_t)MPAT * KPAT;
    patch_gather_kernel<<<(unsigned)((n + 255) / 256), 256, 0, stream>>>(img, apat);
  }
  launch_gemm<0>(apat, wpatcht, bpatch, ptmp, nullptr, MPAT, DMODEL, KPAT, stream);
  {
    size_t n = (size_t)MTOK * DMODEL;
    assemble_tok_kernel<<<(unsigned)((n + 255) / 256), 256, 0, stream>>>(
        ptmp, cls, pos, x, xbf);
  }

  // --- transformer layers ---
  for (int l = 0; l < NLAYER; ++l) {
    launch_gemm<0>(xbf, wqkvt + (size_t)l * 3 * DMODEL * DMODEL, nullptr,
                   nullptr, qkvbf, MTOK, 3 * DMODEL, DMODEL, stream);
    {
      size_t n = BH * TPAD * DH;
      split_qkv_kernel<<<(unsigned)((n + 255) / 256), 256, 0, stream>>>(
          qkvbf, Qp, Kp, Vp);
    }
    {
      int waves = (int)BH * QTILES;  // 4736, divisible by 8
      flash_attn_kernel<<<waves / 8, 256, 0, stream>>>(Qp, Kp, Vp, obf);
    }
    launch_gemm<0>(obf, w0t + (size_t)l * DMODEL * DMODEL, nullptr, gf32,
                   nullptr, MTOK, DMODEL, DMODEL, stream);
    add_ln_kernel<<<MTOK, 256, 0, stream>>>(gf32, x, g1 + l * DMODEL,
                                            b1 + l * DMODEL, y, ybf);
    launch_gemm<1>(ybf, w1t + (size_t)l * DFF * DMODEL, bb1 + l * DFF, nullptr,
                   hbf, MTOK, DFF, DMODEL, stream);
    launch_gemm<0>(hbf, w2t + (size_t)l * DMODEL * DFF, bb2 + l * DMODEL, gf32,
                   nullptr, MTOK, DMODEL, DFF, stream);
    add_ln_kernel<<<MTOK, 256, 0, stream>>>(gf32, y, g2 + l * DMODEL,
                                            b2 + l * DMODEL, x, xbf);
  }

  // --- classifier head ---
  head_kernel<<<1, 320, 0, stream>>>(x, Whead, bhead, out);
}
